// Segger_60636348285162
// MI455X (gfx1250) — compile-verified, matrix operand order checked by reading, not run on hardware
//
#include <hip/hip_runtime.h>
#include <hip/hip_bf16.h>
#include <math.h>

// GATv2 GNN for MI455X (gfx1250, wave32, WMMA).
// Dense projections run as bf16 WMMA GEMMs (v_wmma_f32_16x16x32_bf16, f32 acc);
// edge softmax/aggregation is the memory-bound scatter/gather part.

#define HEADS 4
#define CH    32
#define HC    128

typedef __attribute__((ext_vector_type(16))) __bf16 v16bf;
typedef __attribute__((ext_vector_type(8)))  float  v8f;
typedef __attribute__((ext_vector_type(4)))  unsigned int v4u;
typedef __attribute__((ext_vector_type(4)))  float  v4f;

union FragU { v4u u4[2]; v16bf v; };   // 32 bytes = 8 VGPRs of bf16 fragment

__device__ __forceinline__ unsigned short f32_to_bf16_bits(float f) {
  unsigned int u = __float_as_uint(f);
  u += 0x7fffu + ((u >> 16) & 1u);     // round-to-nearest-even
  return (unsigned short)(u >> 16);
}

__device__ __forceinline__ void atomicMaxF(float* addr, float val) {
  unsigned int* ua = (unsigned int*)addr;
  unsigned int old = *ua;
  while (__uint_as_float(old) < val) {
    unsigned int assumed = old;
    old = atomicCAS(ua, assumed, __float_as_uint(val));
    if (old == assumed) break;
  }
}

// C[M x Nc](f32 or bf16) = act(A[M x K](bf16,row-major,lda) @ B[K x Nc](f32) + bias)
// One wave computes one 16x16 tile; 8 waves per block share one 16-col B panel,
// pre-swizzled into WMMA B-fragment order in LDS (bf16).
__global__ void __launch_bounds__(256)
wmma_gemm_kernel(const unsigned short* __restrict__ A, int lda,
                 const float* __restrict__ B, int ldb,
                 const float* __restrict__ bias,
                 float* __restrict__ Cf, unsigned short* __restrict__ Cb, int ldc,
                 int Mtiles, int K, int relu)
{
  __shared__ unsigned short ldsB[128 * 16];   // up to K=128, fragment-ordered
  const int n0   = blockIdx.x * 16;
  const int lane = threadIdx.x & 31;
  const int wave = threadIdx.x >> 5;
  const int nk   = K >> 5;                    // K in chunks of 32

  // Cooperative fill: element e of lane l's fragment for k-chunk kc holds
  // B[kc*32 + kbase + kin, n0 + (l&15)] with kbase=(l>>4)*8,
  // kin = {0..7} for elems 0..7 and {16..23} for elems 8..15 (16-bit A/B layout).
  for (int idx = threadIdx.x; idx < nk * 512; idx += 256) {
    int kc = idx >> 9;
    int l  = (idx >> 4) & 31;
    int e  = idx & 15;
    int vg = e >> 1, hf = e & 1;
    int kin = (vg < 4) ? (2 * vg + hf) : (16 + 2 * (vg - 4) + hf);
    int k = kc * 32 + ((l >> 4) << 3) + kin;
    ldsB[idx] = f32_to_bf16_bits(B[(size_t)k * ldb + n0 + (l & 15)]);
  }
  __syncthreads();

  const int mtile = blockIdx.y * 8 + wave;
  if (mtile >= Mtiles) return;
  const int m0 = mtile << 4;

  v8f acc = {};
  const unsigned short* abase =
      A + (size_t)(m0 + (lane & 15)) * lda + ((lane >> 4) << 3);

  for (int kc = 0; kc < nk; ++kc) {
    __builtin_prefetch(abase + (kc + 1) * 32, 0, 0);   // global_prefetch_b8
    FragU fa, fb;
    const v4u* ap = (const v4u*)(abase + kc * 32);
    fa.u4[0] = ap[0];                                  // K = kbase + 0..7
    fa.u4[1] = ap[2];                                  // K = kbase + 16..23
    const v4u* bp = (const v4u*)&ldsB[(kc * 32 + lane) * 16];
    fb.u4[0] = bp[0];
    fb.u4[1] = bp[1];
    acc = __builtin_amdgcn_wmma_f32_16x16x32_bf16(false, fa.v, false, fb.v,
                                                  (short)0, acc, false, false);
  }

  const int   col   = n0 + (lane & 15);
  const float bv    = bias ? bias[col] : 0.0f;
  const int   rbase = m0 + ((lane >> 4) << 3);   // C layout: lanes16-31 -> M+8
  #pragma unroll
  for (int r = 0; r < 8; ++r) {
    float v = acc[r] + bv;
    if (relu) v = fmaxf(v, 0.0f);
    size_t o = (size_t)(rbase + r) * ldc + col;
    if (Cb) Cb[o] = f32_to_bf16_bits(v);
    else    Cf[o] = v;
  }
}

__global__ void cvt_f32_to_bf16_kernel(const float* __restrict__ in,
                                       unsigned short* __restrict__ out, int n) {
  int i = blockIdx.x * blockDim.x + threadIdx.x;
  if (i < n) out[i] = f32_to_bf16_bits(in[i]);
}

__global__ void seg_init_kernel(float* __restrict__ mmax, float* __restrict__ den,
                                float* __restrict__ aggr, int nH, int nHC) {
  int i = blockIdx.x * blockDim.x + threadIdx.x;
  if (i < nHC) aggr[i] = 0.0f;
  if (i < nH) { mmax[i] = -__builtin_inff(); den[i] = 0.0f; }
}

// score[e,h] = sum_c att[h,c] * leakyrelu(xl[src,h,c] + xr[dst,h,c]); seg-max via CAS
__global__ void edge_score_kernel(const float* __restrict__ xl, const float* __restrict__ xr,
                                  const int* __restrict__ src, const int* __restrict__ dst,
                                  const float* __restrict__ att,
                                  float* __restrict__ score, float* __restrict__ mmax, int E) {
  int t = blockIdx.x * blockDim.x + threadIdx.x;
  if (t >= E * HEADS) return;
  int e = t >> 2, h = t & 3;
  const v4f* pl = (const v4f*)(xl + (size_t)src[e] * HC + h * CH);
  const v4f* pr = (const v4f*)(xr + (size_t)dst[e] * HC + h * CH);
  const v4f* pa = (const v4f*)(att + h * CH);
  float s = 0.0f;
  #pragma unroll
  for (int c = 0; c < CH / 4; ++c) {
    v4f a = pl[c], b = pr[c], w = pa[c];
    #pragma unroll
    for (int j = 0; j < 4; ++j) {
      float ev = a[j] + b[j];
      s += w[j] * (ev > 0.0f ? ev : 0.2f * ev);
    }
  }
  score[t] = s;
  atomicMaxF(&mmax[(size_t)dst[e] * HEADS + h], s);
}

__global__ void edge_exp_kernel(const float* __restrict__ score, const int* __restrict__ dst,
                                const float* __restrict__ mmax,
                                float* __restrict__ exv, float* __restrict__ den, int E) {
  int t = blockIdx.x * blockDim.x + threadIdx.x;
  if (t >= E * HEADS) return;
  int e = t >> 2, h = t & 3;
  int d = dst[e];
  float ex = __expf(score[t] - mmax[(size_t)d * HEADS + h]);
  exv[t] = ex;
  atomicAdd(&den[(size_t)d * HEADS + h], ex);
}

__global__ void edge_aggr_kernel(const float* __restrict__ xl,
                                 const int* __restrict__ src, const int* __restrict__ dst,
                                 const float* __restrict__ exv,
                                 float* __restrict__ aggr, int E) {
  int t = blockIdx.x * blockDim.x + threadIdx.x;
  if (t >= E * HEADS) return;
  int e = t >> 2, h = t & 3;
  float ex = exv[t];
  const float* p = xl + (size_t)src[e] * HC + h * CH;
  float* q = aggr + (size_t)dst[e] * HC + h * CH;
  #pragma unroll
  for (int c = 0; c < CH; ++c) atomicAdd(&q[c], p[c] * ex);
}

// h_next = act(aggr/den + concat_bias + skip); bf16 out (next layer) or f32 (last)
__global__ void finalize_kernel(const float* __restrict__ aggr, const float* __restrict__ den,
                                const float* __restrict__ skip, const float* __restrict__ cb,
                                unsigned short* __restrict__ hout_b, float* __restrict__ hout_f,
                                int relu, int nHC) {
  int i = blockIdx.x * blockDim.x + threadIdx.x;
  if (i >= nHC) return;
  int j = i & (HC - 1);
  int n = i >> 7;
  float d = den[(size_t)n * HEADS + (j >> 5)];
  float g = (d > 0.0f) ? aggr[i] / d : 0.0f;
  float v = g + cb[j] + skip[i];
  if (relu) v = fmaxf(v, 0.0f);
  if (hout_b) hout_b[i] = f32_to_bf16_bits(v);
  if (hout_f) hout_f[i] = v;
}

__global__ void normalize_kernel(const float* __restrict__ hin, float* __restrict__ out, int n) {
  int i = blockIdx.x * blockDim.x + threadIdx.x;
  if (i >= n) return;
  const float* p = hin + (size_t)i * HC;
  float ss = 0.0f;
  #pragma unroll 4
  for (int j = 0; j < HC; ++j) ss += p[j] * p[j];
  float inv = rsqrtf(ss);
  #pragma unroll 4
  for (int j = 0; j < HC; ++j) out[(size_t)i * HC + j] = p[j] * inv;
}

extern "C" void kernel_launch(void* const* d_in, const int* in_sizes, int n_in,
                              void* d_out, int out_size, void* d_ws, size_t ws_size,
                              hipStream_t stream) {
  const float* x  = (const float*)d_in[0];
  const int*   ei = (const int*)d_in[1];
  const float* W0 = (const float*)d_in[2];
  const int N = in_sizes[0] / 64;
  const int E = in_sizes[1] / 2;
  const int* src = ei;
  const int* dst = ei + E;

  // Workspace carve-up (~280 MB)
  char* ws = (char*)d_ws;
  size_t off = 0;
  auto carve = [&](size_t bytes) -> void* {
    void* p = ws + off; off += (bytes + 255) & ~(size_t)255; return p;
  };
  unsigned short* xb   = (unsigned short*)carve((size_t)N * 64 * 2);
  unsigned short* h    = (unsigned short*)carve((size_t)N * HC * 2);
  float* xl    = (float*)carve((size_t)N * HC * 4);
  float* xr    = (float*)carve((size_t)N * HC * 4);
  float* skip  = (float*)carve((size_t)N * HC * 4);
  float* aggr  = (float*)carve((size_t)N * HC * 4);
  float* score = (float*)carve((size_t)E * HEADS * 4);
  float* exv   = (float*)carve((size_t)E * HEADS * 4);
  float* mmax  = (float*)carve((size_t)N * HEADS * 4);
  float* den   = (float*)carve((size_t)N * HEADS * 4);
  (void)ws_size; (void)n_in;

  const int BLK = 256;
  const int Mtiles = N / 16;                 // 100000/16 = 6250 exactly
  dim3 gblk(256);
  auto ggrid = [&](int Nc) { return dim3(Nc / 16, (Mtiles + 7) / 8); };
  const int nHC = N * HC, nH = N * HEADS, nEH = E * HEADS;

  // h0 = relu(x @ W0), written bf16 with ld=32
  int nx = N * 64;
  cvt_f32_to_bf16_kernel<<<(nx + BLK - 1) / BLK, BLK, 0, stream>>>(x, xb, nx);
  wmma_gemm_kernel<<<ggrid(32), gblk, 0, stream>>>(
      xb, 64, W0, 32, (const float*)nullptr,
      (float*)nullptr, h, 32, Mtiles, 64, 1);

  auto run_layer = [&](int base, int din, bool last) {
    const float* Wl  = (const float*)d_in[base + 0];
    const float* bl  = (const float*)d_in[base + 1];
    const float* Wr  = (const float*)d_in[base + 2];
    const float* br  = (const float*)d_in[base + 3];
    const float* att = (const float*)d_in[base + 4];
    const float* cb  = (const float*)d_in[base + 5];
    const float* Wsk = (const float*)d_in[base + 6];
    const float* bsk = (const float*)d_in[base + 7];

    wmma_gemm_kernel<<<ggrid(HC), gblk, 0, stream>>>(
        h, din, Wl, HC, bl, xl, (unsigned short*)nullptr, HC, Mtiles, din, 0);
    wmma_gemm_kernel<<<ggrid(HC), gblk, 0, stream>>>(
        h, din, Wr, HC, br, xr, (unsigned short*)nullptr, HC, Mtiles, din, 0);
    wmma_gemm_kernel<<<ggrid(HC), gblk, 0, stream>>>(
        h, din, Wsk, HC, bsk, skip, (unsigned short*)nullptr, HC, Mtiles, din, 0);

    seg_init_kernel<<<(nHC + BLK - 1) / BLK, BLK, 0, stream>>>(mmax, den, aggr, nH, nHC);
    edge_score_kernel<<<(nEH + BLK - 1) / BLK, BLK, 0, stream>>>(xl, xr, src, dst, att, score, mmax, E);
    edge_exp_kernel<<<(nEH + BLK - 1) / BLK, BLK, 0, stream>>>(score, dst, mmax, exv, den, E);
    edge_aggr_kernel<<<(nEH + BLK - 1) / BLK, BLK, 0, stream>>>(xl, src, dst, exv, aggr, E);
    finalize_kernel<<<(nHC + BLK - 1) / BLK, BLK, 0, stream>>>(
        aggr, den, skip, cb,
        last ? (unsigned short*)nullptr : h,
        last ? xl : (float*)nullptr,          // xl consumed; reuse as last-layer f32 h
        last ? 0 : 1, nHC);
  };

  run_layer(3, 32, false);      // conv_first (din = EMB = 32)
  run_layer(11, HC, false);     // conv_mid x3 (shared weights)
  run_layer(11, HC, false);
  run_layer(11, HC, false);
  run_layer(19, HC, true);      // conv_last -> f32 into xl

  normalize_kernel<<<(N + BLK - 1) / BLK, BLK, 0, stream>>>(xl, (float*)d_out, N);
}
